// MultiHeadSelfAttentionNoRoPE_31714038513758
// MI455X (gfx1250) — matrix operand order repaired and checked by reading, not compiled, not run on hardware
//
#include <hip/hip_runtime.h>
#include <hip/hip_bf16.h>

// ---------------------------------------------------------------------------
// MI455X (gfx1250) MHA forward, bf16 WMMA pipeline.
//   x:[4,2048,1024] f32, w_{q,k,v,o}:[1024,1024] f32 (y = x @ W^T)
//   1) convert x + weights to bf16 (fp32 exponent range; problem is matrix-
//      compute bound -> v_wmma_f32_16x16x32_bf16 path)
//   2) Q,K -> bf16 [B,H,S,64]; V -> bf16 [B,H,64,S] (transposed so PV's WMMA
//      B-fragment is two contiguous 8-elem loads)
//   3) causal flash-attention: WMMA QK^T + PV, K/V^T tiles staged via async
//      global->LDS DMA (ASYNCcnt) into double-buffered LDS, probs bounced
//      through a wave-private LDS tile for C-layout -> A-layout
//   4) out = ctx Wo^T -> f32 d_out
// GEMMs are software-pipelined (ping-pong fragment registers) so WMMAs never
// wait on same-iteration loads. Workspace: ~88 MB.
// ---------------------------------------------------------------------------

typedef __attribute__((ext_vector_type(16))) __bf16 v16bf;
typedef __attribute__((ext_vector_type(8)))  __bf16 v8bf;
typedef __attribute__((ext_vector_type(8)))  float  v8f;

union ABfrag { v16bf v; v8bf h[2]; };

#define D_MODEL  1024
#define N_HEADS  16
#define HEAD_DIM 64
#define BATCH    4
#define SEQ      2048
#define ROWS     (BATCH * SEQ)   // 8192

#if defined(__has_builtin)
#if __has_builtin(__builtin_amdgcn_global_load_async_to_lds_b128)
#define HAS_ASYNC 1
#endif
#endif
#ifndef HAS_ASYNC
#define HAS_ASYNC 0
#endif

#if HAS_ASYNC
// Builtin signature (from clang diagnostics): params are pointers to a
// 16-byte int vector, addrspace(1) for global and addrspace(3) for LDS,
// plus imm offset and imm cpol.
typedef int async_v4i __attribute__((vector_size(16)));
typedef __attribute__((address_space(1))) async_v4i* as1_v4i_ptr;
typedef __attribute__((address_space(3))) async_v4i* as3_v4i_ptr;
#define ASYNC_COPY16(gsrc, ldst)                                               \
  __builtin_amdgcn_global_load_async_to_lds_b128(                              \
      (as1_v4i_ptr)(gsrc), (as3_v4i_ptr)(ldst), 0, 0)
#if __has_builtin(__builtin_amdgcn_s_wait_asynccnt)
#define WAIT_ASYNC(n) __builtin_amdgcn_s_wait_asynccnt(n)
#else
#define WAIT_ASYNC(n) asm volatile("s_wait_asynccnt " #n ::: "memory")
#endif
#endif

__device__ __forceinline__ v8f wmma_bf16(const ABfrag& a, const ABfrag& b, v8f c) {
  return __builtin_amdgcn_wmma_f32_16x16x32_bf16(
      /*neg_a=*/false, a.v, /*neg_b=*/false, b.v,
      /*c_mod=*/(short)0, c, /*reuse_a=*/false, /*reuse_b=*/false);
}

__device__ __forceinline__ float rmax16(float x) {
#pragma unroll
  for (int m = 1; m < 16; m <<= 1) x = fmaxf(x, __shfl_xor(x, m, 32));
  return x;
}
__device__ __forceinline__ float rsum16(float x) {
#pragma unroll
  for (int m = 1; m < 16; m <<= 1) x += __shfl_xor(x, m, 32);
  return x;
}

// ---------------------------------------------------------------------------
__global__ void cvt_f32_bf16(const float* __restrict__ src,
                             __bf16* __restrict__ dst, int n) {
  int i = blockIdx.x * blockDim.x + threadIdx.x;
  int stride = gridDim.x * blockDim.x;
  for (; i < n; i += stride) dst[i] = (__bf16)src[i];
}

// ---------------------------------------------------------------------------
// C[M,N] = A[M,K] * W[N,K]^T, bf16 in, fp32 accum.
// Block = 256 threads = 8 waves in 2(M) x 4(N); wave tile 32M x 64N
// (2 A-frags x 4 B-frags = 8 WMMAs per 32-deep k-step, 12 b128 loads).
// k-loop is ping-pong double buffered: WMMAs consume fragments loaded one
// iteration earlier, so s_wait_loadcnt gets a full iteration of slack.
// mode 0: bf16 out scattered to [B,H,S,64]   (Q, K)
// mode 1: bf16 out scattered to [B,H,64,S]   (V transposed)
// mode 2: f32  out row-major [M,N]           (final projection)
__global__ void __launch_bounds__(256)
gemm_proj(const __bf16* __restrict__ A, const __bf16* __restrict__ W,
          void* __restrict__ out, int mode) {
  const int lane = threadIdx.x & 31;
  const int wid  = threadIdx.x >> 5;
  const int wm = wid >> 2, wn = wid & 3;
  const int mt = blockIdx.x * 64 + wm * 32;
  const int nb = blockIdx.y * 256 + wn * 64;
  const int l15 = lane & 15;
  const int hi  = lane >> 4;

  const __bf16* Ap0 = A + (mt + l15) * D_MODEL + (hi ? 8 : 0);   // rows mt..mt+15
  const __bf16* Ap1 = Ap0 + 16 * D_MODEL;                        // rows mt+16..mt+31
  const __bf16* Wp0 = W + (nb + l15) * D_MODEL + (hi ? 8 : 0);

  ABfrag a0[2], b0[4], a1[2], b1[4];
  auto load_k = [&](int kb, ABfrag* af, ABfrag* bf) {
    af[0].h[0] = *(const v8bf*)(Ap0 + kb);
    af[0].h[1] = *(const v8bf*)(Ap0 + kb + 16);
    af[1].h[0] = *(const v8bf*)(Ap1 + kb);
    af[1].h[1] = *(const v8bf*)(Ap1 + kb + 16);
#pragma unroll
    for (int j = 0; j < 4; ++j) {
      const __bf16* wp = Wp0 + j * 16 * D_MODEL + kb;
      bf[j].h[0] = *(const v8bf*)(wp);
      bf[j].h[1] = *(const v8bf*)(wp + 16);
    }
  };
  v8f c[2][4] = {};
  auto mm = [&](ABfrag* af, ABfrag* bf) {
#pragma unroll
    for (int j = 0; j < 4; ++j) {
      c[0][j] = wmma_bf16(af[0], bf[j], c[0][j]);
      c[1][j] = wmma_bf16(af[1], bf[j], c[1][j]);
    }
  };

  load_k(0, a0, b0);
  for (int kb = 0; kb < D_MODEL; kb += 64) {
    __builtin_prefetch(Ap0 + kb + 128, 0, 3);
    __builtin_prefetch(Ap1 + kb + 128, 0, 3);
    load_k(kb + 32, a1, b1);     // next half-step DMA in flight
    mm(a0, b0);                  // consume fragments loaded last step
    if (kb + 64 < D_MODEL) load_k(kb + 64, a0, b0);
    mm(a1, b1);
  }

#pragma unroll
  for (int mm2 = 0; mm2 < 2; ++mm2)
#pragma unroll
    for (int j = 0; j < 4; ++j)
#pragma unroll
      for (int r = 0; r < 8; ++r) {
        float v = c[mm2][j][r];
        int row = mt + mm2 * 16 + r + hi * 8;   // C layout: M = r + 8*(lane>=16)
        int col = nb + j * 16 + l15;            //           N = lane & 15
        if (mode == 2) {
          ((float*)out)[row * D_MODEL + col] = v;
        } else {
          int bb = row >> 11, s = row & (SEQ - 1);
          int hh = col >> 6,  d = col & (HEAD_DIM - 1);
          if (mode == 0)
            ((__bf16*)out)[((bb * N_HEADS + hh) * SEQ + s) * HEAD_DIM + d] = (__bf16)v;
          else
            ((__bf16*)out)[((bb * N_HEADS + hh) * HEAD_DIM + d) * SEQ + s] = (__bf16)v;
        }
      }
}

// ---------------------------------------------------------------------------
// Causal flash attention. Block = (b, h, 128 q-rows), 8 waves x 16 q-rows.
// K/V^T tiles (32 keys) double-buffered in LDS, filled by async global->LDS
// DMA: tile kt+1's transfer overlaps tile kt's 8 WMMAs + softmax.
__global__ void __launch_bounds__(256)
attn_kernel(const __bf16* __restrict__ Q, const __bf16* __restrict__ K,
            const __bf16* __restrict__ Vt, __bf16* __restrict__ ctx) {
  __shared__ __bf16 Kl[2][32 * 64];    // [buf][key][d]
  __shared__ __bf16 Vl[2][64 * 32];    // [buf][d][key]
  __shared__ __bf16 Pl[8][16 * 32];    // per-wave probs bounce [m][key]

  const int t    = threadIdx.x;
  const int lane = t & 31;
  const int wid  = t >> 5;
  const int l15 = lane & 15;
  const int hi  = lane >> 4;
  const int qb = blockIdx.x * 128;
  const int h  = blockIdx.y;
  const int b  = blockIdx.z;
  const int qw = qb + wid * 16;        // wave's first q row

  const __bf16* Qh = Q  + (size_t)(b * N_HEADS + h) * SEQ * HEAD_DIM;
  const __bf16* Kh = K  + (size_t)(b * N_HEADS + h) * SEQ * HEAD_DIM;
  const __bf16* Vh = Vt + (size_t)(b * N_HEADS + h) * HEAD_DIM * SEQ;

  // staging pattern: each thread moves one b128 of K and one of V^T per tile
  const int skey = t >> 3, sdd = (t & 7) * 8;   // K tile  [32][64]
  const int sd2  = t >> 2, sk8 = (t & 3) * 8;   // V^T tile [64][32]
  auto stage = [&](int kt, int bufi) {
    const __bf16* gk = Kh + (kt * 32 + skey) * HEAD_DIM + sdd;
    const __bf16* gv = Vh + sd2 * SEQ + kt * 32 + sk8;
#if HAS_ASYNC
    ASYNC_COPY16(gk, &Kl[bufi][skey * 64 + sdd]);
    ASYNC_COPY16(gv, &Vl[bufi][sd2 * 32 + sk8]);
#else
    *(v8bf*)&Kl[bufi][skey * 64 + sdd] = *(const v8bf*)gk;
    *(v8bf*)&Vl[bufi][sd2 * 32 + sk8]  = *(const v8bf*)gv;
#endif
  };

  ABfrag qa[2];                         // Q rows, two 32-wide d chunks
  {
    const __bf16* qp = Qh + (qw + l15) * HEAD_DIM + (hi ? 8 : 0);
#pragma unroll
    for (int kc = 0; kc < 2; ++kc) {
      qa[kc].h[0] = *(const v8bf*)(qp + kc * 32);
      qa[kc].h[1] = *(const v8bf*)(qp + kc * 32 + 16);
    }
  }

  v8f acc[4] = {};
  float mr[8], lr[8];
#pragma unroll
  for (int r = 0; r < 8; ++r) { mr[r] = -__builtin_inff(); lr[r] = 0.f; }

  const int ntiles = (qb + 128) / 32;  // block-level causal bound
  const int qmaxw  = qw + 15;

  stage(0, 0);
  for (int kt = 0; kt < ntiles; ++kt) {
    const int kb = kt * 32;
    const int cur = kt & 1;
    __syncthreads();                   // compute kt-1 done -> buf[(kt+1)&1] free
    const bool more = (kt + 1 < ntiles);
    if (more) stage(kt + 1, cur ^ 1);  // next tile DMA overlaps this compute
#if HAS_ASYNC
    if (more) { WAIT_ASYNC(2); } else { WAIT_ASYNC(0); }  // my tile-kt done
#endif
    __syncthreads();                   // tile kt visible block-wide
    if (kb > qmaxw) continue;          // fully-masked tile for this wave

    v8f sf[2] = {};                    // scores: two 16-key subtiles
#pragma unroll
    for (int sub = 0; sub < 2; ++sub) {
#pragma unroll
      for (int kc = 0; kc < 2; ++kc) {
        const __bf16* kp = &Kl[cur][(sub * 16 + l15) * 64 + kc * 32 + (hi ? 8 : 0)];
        ABfrag bf;
        bf.h[0] = *(const v8bf*)kp;
        bf.h[1] = *(const v8bf*)(kp + 16);
        sf[sub] = wmma_bf16(qa[kc], bf, sf[sub]);
      }
    }

    float fac[8];
#pragma unroll
    for (int r = 0; r < 8; ++r) {      // scale, mask, online softmax
      int q = qw + r + hi * 8;
      float s0 = (kb + l15      <= q) ? sf[0][r] * 0.125f : -__builtin_inff();
      float s1 = (kb + 16 + l15 <= q) ? sf[1][r] * 0.125f : -__builtin_inff();
      float tm = rmax16(fmaxf(s0, s1));
      float mn = fmaxf(mr[r], tm);
      fac[r] = __expf(mr[r] - mn);
      mr[r] = mn;
      float p0 = __expf(s0 - mn);
      float p1 = __expf(s1 - mn);
      int m = r + hi * 8;
      Pl[wid][m * 32 + l15]      = (__bf16)p0;   // C-layout -> LDS row-major
      Pl[wid][m * 32 + 16 + l15] = (__bf16)p1;
      lr[r] = lr[r] * fac[r] + rsum16(p0 + p1);
    }
#pragma unroll
    for (int ds = 0; ds < 4; ++ds)
#pragma unroll
      for (int r = 0; r < 8; ++r) acc[ds][r] *= fac[r];

    ABfrag pa;                         // reload probs in A-layout
    {
      const __bf16* pp = &Pl[wid][l15 * 32 + (hi ? 8 : 0)];
      pa.h[0] = *(const v8bf*)(pp);
      pa.h[1] = *(const v8bf*)(pp + 16);
    }
#pragma unroll
    for (int ds = 0; ds < 4; ++ds) {   // PV: contraction over the 32 keys
      const __bf16* vp = &Vl[cur][(ds * 16 + l15) * 32 + (hi ? 8 : 0)];
      ABfrag vb;
      vb.h[0] = *(const v8bf*)vp;
      vb.h[1] = *(const v8bf*)(vp + 16);
      acc[ds] = wmma_bf16(pa, vb, acc[ds]);
    }
  }

#pragma unroll
  for (int ds = 0; ds < 4; ++ds)       // ctx[b,s,h*64+d] bf16
#pragma unroll
    for (int r = 0; r < 8; ++r) {
      int s = qw + r + hi * 8;
      int d = ds * 16 + l15;
      float v = acc[ds][r] / lr[r];
      ctx[(size_t)(b * SEQ + s) * D_MODEL + h * HEAD_DIM + d] = (__bf16)v;
    }
}

// ---------------------------------------------------------------------------
extern "C" void kernel_launch(void* const* d_in, const int* in_sizes, int n_in,
                              void* d_out, int out_size, void* d_ws, size_t ws_size,
                              hipStream_t stream) {
  const float* x  = (const float*)d_in[0];
  const float* wq = (const float*)d_in[1];
  const float* wk = (const float*)d_in[2];
  const float* wv = (const float*)d_in[3];
  const float* wo = (const float*)d_in[4];

  char* ws = (char*)d_ws;
  size_t off = 0;
  auto carve = [&](size_t bytes) -> void* {
    void* p = ws + off;
    off += (bytes + 255) & ~(size_t)255;
    return p;
  };
  __bf16* xbf = (__bf16*)carve((size_t)ROWS * D_MODEL * 2);
  __bf16* wqb = (__bf16*)carve((size_t)D_MODEL * D_MODEL * 2);
  __bf16* wkb = (__bf16*)carve((size_t)D_MODEL * D_MODEL * 2);
  __bf16* wvb = (__bf16*)carve((size_t)D_MODEL * D_MODEL * 2);
  __bf16* wob = (__bf16*)carve((size_t)D_MODEL * D_MODEL * 2);
  __bf16* Qb  = (__bf16*)carve((size_t)ROWS * D_MODEL * 2);
  __bf16* Kb  = (__bf16*)carve((size_t)ROWS * D_MODEL * 2);
  __bf16* Vtb = (__bf16*)carve((size_t)ROWS * D_MODEL * 2);
  __bf16* ctx = (__bf16*)carve((size_t)ROWS * D_MODEL * 2);

  cvt_f32_bf16<<<512, 256, 0, stream>>>(x,  xbf, ROWS * D_MODEL);
  cvt_f32_bf16<<<256, 256, 0, stream>>>(wq, wqb, D_MODEL * D_MODEL);
  cvt_f32_bf16<<<256, 256, 0, stream>>>(wk, wkb, D_MODEL * D_MODEL);
  cvt_f32_bf16<<<256, 256, 0, stream>>>(wv, wvb, D_MODEL * D_MODEL);
  cvt_f32_bf16<<<256, 256, 0, stream>>>(wo, wob, D_MODEL * D_MODEL);

  dim3 gg(ROWS / 64, D_MODEL / 256), blk(256);
  gemm_proj<<<gg, blk, 0, stream>>>(xbf, wqb, Qb, 0);
  gemm_proj<<<gg, blk, 0, stream>>>(xbf, wkb, Kb, 0);
  gemm_proj<<<gg, blk, 0, stream>>>(xbf, wvb, Vtb, 1);

  attn_kernel<<<dim3(SEQ / 128, N_HEADS, BATCH), blk, 0, stream>>>(Qb, Kb, Vtb, ctx);

  gemm_proj<<<gg, blk, 0, stream>>>(ctx, wob, d_out, 2);
}
